// ImageGenerator_31774168056054
// MI455X (gfx1250) — compile-verified, matrix operand order checked by reading, not run on hardware
//
#include <hip/hip_runtime.h>
#include <hip/hip_bf16.h>
#include <stdint.h>

// ---------------- problem constants ----------------
#define TN   32768          // B*N nodes
#define EN   262144         // edges
#define NPTS 8192
#define BB   4

typedef __attribute__((ext_vector_type(16))) __bf16 v16bf;
typedef __attribute__((ext_vector_type(8)))  __bf16 v8bf;
typedef __attribute__((ext_vector_type(8)))  float  v8f;

union ABfrag { v16bf v; v8bf h[2]; };

__device__ __forceinline__ __bf16 f2bf(float f) {
  unsigned u = __builtin_bit_cast(unsigned, f);
  unsigned r = (u + 0x7FFFu + ((u >> 16) & 1u)) >> 16;
  unsigned short s = (unsigned short)r;
  return __builtin_bit_cast(__bf16, s);
}
__device__ __forceinline__ float bf2f(__bf16 x) {
  unsigned short s = __builtin_bit_cast(unsigned short, x);
  unsigned u = ((unsigned)s) << 16;
  return __builtin_bit_cast(float, u);
}
__device__ __forceinline__ float act_apply(float v, int act) {
  if (act == 1) return fmaxf(v, 0.0f);
  if (act == 2) return v > 0.0f ? v : 0.01f * v;   // jax leaky default
  if (act == 3) return v > 0.0f ? v : 0.2f * v;
  return v;
}

// ---------------- fp32 -> padded bf16 weight convert ----------------
__global__ void __launch_bounds__(256)
cvt_w(const float* __restrict__ W, __bf16* __restrict__ dst,
      int rows, int kin, int kpad) {
  int i = blockIdx.x * 256 + threadIdx.x;
  if (i >= rows * kpad) return;
  int r = i / kpad, c = i - r * kpad;
  dst[i] = (c < kin) ? f2bf(W[(size_t)r * kin + c]) : f2bf(0.0f);
}

// ---- f-weight convert with edge-layout remap: cols 0..2 geo, 32..159 feat ----
__global__ void __launch_bounds__(256)
cvt_wf(const float* __restrict__ W, __bf16* __restrict__ dst) {
  int i = blockIdx.x * 256 + threadIdx.x;
  if (i >= 128 * 160) return;
  int r = i / 160, c = i - r * 160;
  float v = 0.0f;
  if (c < 3)        v = W[(size_t)r * 131 + c];
  else if (c >= 32) v = W[(size_t)r * 131 + (c - 29)];
  dst[i] = f2bf(v);
}

// ---------------- pack [zf | pos3] -> bf16 [T,160] ----------------
__global__ void __launch_bounds__(256)
pack160(const float* __restrict__ z, const float* __restrict__ p3,
        __bf16* __restrict__ X) {
  int i = blockIdx.x * 256 + threadIdx.x;
  if (i >= TN * 160) return;
  int t = i / 160, c = i - t * 160;
  float v;
  if (c < 128)      v = z[(size_t)t * 128 + c];
  else if (c < 131) v = p3[(size_t)t * 3 + (c - 128)];
  else              v = 0.0f;
  X[i] = f2bf(v);
}

// ---------------- RFF encoding ----------------
__global__ void __launch_bounds__(256)
rff_k(const float* __restrict__ pos, const float* __restrict__ encb,
      float* __restrict__ Hf, __bf16* __restrict__ Hb) {
  int i = blockIdx.x * 256 + threadIdx.x;
  if (i >= TN * 64) return;
  int t = i / 64, j = i - t * 64;
  float v = 6.2831853071795864769f *
            (pos[t * 3 + 0] * encb[j * 3 + 0] +
             pos[t * 3 + 1] * encb[j * 3 + 1] +
             pos[t * 3 + 2] * encb[j * 3 + 2]);
  float c = cosf(v), s = sinf(v);
  size_t o = (size_t)t * 128;
  Hf[o + j] = c;       Hf[o + 64 + j] = s;
  Hb[o + j] = f2bf(c); Hb[o + 64 + j] = f2bf(s);
}

// ---------------- WMMA GEMM: Y = act(X[T,ldx] @ W^T + b) ----------------
// One wave owns a 16x64 strip: A fragment loaded once per K-step, reused
// across 4 column tiles (4 wmma per A load).
__global__ void __launch_bounds__(256)
gemm_wmma(const __bf16* __restrict__ X, int ldx,
          const __bf16* __restrict__ W,
          const float* __restrict__ bias,
          float* __restrict__ Yf, __bf16* __restrict__ Yb,
          int Nout, int act) {
  int lane = threadIdx.x & 31;
  int wv   = threadIdx.x >> 5;
  int n4   = Nout >> 6;                        // groups of 4 N-tiles
  int tile = blockIdx.x * 8 + wv;
  int tg = tile % n4;
  int tm = tile / n4;
  int row  = (tm << 4) + (lane & 15);
  int ncol = lane & 15;
  int ksel = ((lane >> 4) & 1) << 3;

  v8f acc[4] = {};
  for (int kk = 0; kk < ldx; kk += 32) {
    int kb = kk + ksel;
    ABfrag a;
    a.h[0] = *(const v8bf*)(X + (size_t)row * ldx + kb);
    a.h[1] = *(const v8bf*)(X + (size_t)row * ldx + kb + 16);
#pragma unroll
    for (int j = 0; j < 4; j++) {
      int wr = (tg << 6) + (j << 4) + ncol;
      ABfrag b;
      b.h[0] = *(const v8bf*)(W + (size_t)wr * ldx + kb);
      b.h[1] = *(const v8bf*)(W + (size_t)wr * ldx + kb + 16);
      acc[j] = __builtin_amdgcn_wmma_f32_16x16x32_bf16(
          false, a.v, false, b.v, (short)0, acc[j], false, false);
    }
  }
#pragma unroll
  for (int j = 0; j < 4; j++) {
    int wr = (tg << 6) + (j << 4) + ncol;
    float bs = bias ? bias[wr] : 0.0f;
#pragma unroll
    for (int r = 0; r < 8; r++) {
      int m = (tm << 4) + r + ((lane >> 4) << 3);
      float v = act_apply(acc[j][r] + bs, act);
      size_t o = (size_t)m * Nout + wr;
      if (Yf) Yf[o] = v;
      if (Yb) Yb[o] = f2bf(v);
    }
  }
}

// ---------------- delta = tanh(W2 @ relu_h1) : 128 -> 3 ----------------
__global__ void __launch_bounds__(256)
delta_k(const __bf16* __restrict__ T1, const float* __restrict__ W,
        const float* __restrict__ b, float* __restrict__ delta) {
  int t = blockIdx.x * 256 + threadIdx.x;
  if (t >= TN) return;
  for (int d = 0; d < 3; d++) {
    float s = b[d];
    for (int c = 0; c < 128; c++)
      s += bf2f(T1[(size_t)t * 128 + c]) * W[d * 128 + c];
    delta[t * 3 + d] = tanhf(s);
  }
}

// ---------------- fused edge GEMM + segment-sum ----------------
// block = 16 nodes = 128 edges.  Edge row layout (160 cols, bf16):
//   [0..2] = pos_j - pos_i + delta_i, [3..31] = 0, [32..159] = h_j
// Feature block is gathered global->LDS with async b128 copies (ASYNCcnt).
// Each wave owns a 16-row tile (= 2 nodes x 8 edges); A-frag reused across
// all 8 column tiles; relu+bias then 8-register sum = segment_sum.
__global__ void __launch_bounds__(256)
edge_gemm(const __bf16* __restrict__ HB, const float* __restrict__ pos,
          const float* __restrict__ delta, const int* __restrict__ src,
          const __bf16* __restrict__ Wf, const float* __restrict__ bias,
          __bf16* __restrict__ agg) {
  __shared__ __attribute__((aligned(16))) __bf16 elds[128 * 160];
  int tid = threadIdx.x;
  {
    int r = tid >> 1, p = tid & 1;
    int ge = blockIdx.x * 128 + r;
    int s  = src[ge];
    // async gather of the 256B feature row (split across the thread pair)
    unsigned long long gbase =
        (unsigned long long)(uintptr_t)HB + (unsigned long long)s * 256u +
        (unsigned)(p * 128);
    unsigned lbase =
        (unsigned)(uintptr_t)elds + (unsigned)(r * 320 + 64 + p * 128);
#pragma unroll
    for (int c = 0; c < 8; c++) {
      unsigned long long ga = gbase + (unsigned)(c * 16);
      unsigned la = lbase + (unsigned)(c * 16);
      asm volatile("global_load_async_to_lds_b128 %0, %1, off"
                   :: "v"(la), "v"(ga) : "memory");
    }
    if (p == 0) {  // geo + zero pad
      int d = ge >> 3;
      __bf16* row = elds + r * 160;
      row[0] = f2bf(pos[s * 3 + 0] - pos[d * 3 + 0] + delta[d * 3 + 0]);
      row[1] = f2bf(pos[s * 3 + 1] - pos[d * 3 + 1] + delta[d * 3 + 1]);
      row[2] = f2bf(pos[s * 3 + 2] - pos[d * 3 + 2] + delta[d * 3 + 2]);
      for (int c = 3; c < 32; c++) row[c] = f2bf(0.0f);
    }
  }
  asm volatile("s_wait_asynccnt 0" ::: "memory");
  __syncthreads();

  int lane = tid & 31;
  int wv   = tid >> 5;
  int ksel = ((lane >> 4) & 1) << 3;
  int rloc = (wv << 4) + (lane & 15);
  int ncol = lane & 15;
  const __bf16* arow = elds + rloc * 160;

  v8f acc[8] = {};
#pragma unroll
  for (int kk = 0; kk < 160; kk += 32) {
    int kb = kk + ksel;
    ABfrag a;
    a.h[0] = *(const v8bf*)(arow + kb);
    a.h[1] = *(const v8bf*)(arow + kb + 16);
#pragma unroll
    for (int ct = 0; ct < 8; ct++) {
      int wr = (ct << 4) + ncol;
      ABfrag b;
      b.h[0] = *(const v8bf*)(Wf + (size_t)wr * 160 + kb);
      b.h[1] = *(const v8bf*)(Wf + (size_t)wr * 160 + kb + 16);
      acc[ct] = __builtin_amdgcn_wmma_f32_16x16x32_bf16(
          false, a.v, false, b.v, (short)0, acc[ct], false, false);
    }
  }
  int node = blockIdx.x * 16 + (wv << 1) + (lane >> 4);
#pragma unroll
  for (int ct = 0; ct < 8; ct++) {
    int wr = (ct << 4) + ncol;
    float bs = bias[wr];
    float s = 0.0f;
#pragma unroll
    for (int r = 0; r < 8; r++) s += fmaxf(acc[ct][r] + bs, 0.0f);
    agg[(size_t)node * 128 + wr] = f2bf(s);
  }
}

// ---------------- residual + leaky(0.2) ----------------
__global__ void __launch_bounds__(256)
residual_k(const float* __restrict__ H, const float* __restrict__ G2,
           float* __restrict__ HR) {
  int i = blockIdx.x * 256 + threadIdx.x;
  if (i >= TN * 128) return;
  float v = H[i] + G2[i];
  HR[i] = v > 0.0f ? v : 0.2f * v;
}

// ---------------- per-channel mean / rstd over all T nodes ----------------
__global__ void __launch_bounds__(256)
colstats_k(const float* __restrict__ H, float* __restrict__ stats) {
  int c = blockIdx.x;
  float s = 0.0f, ss = 0.0f;
  for (int n = threadIdx.x; n < TN; n += 256) {
    float v = H[(size_t)n * 128 + c];
    s += v; ss += v * v;
  }
  __shared__ float sh0[256], sh1[256];
  sh0[threadIdx.x] = s; sh1[threadIdx.x] = ss;
  __syncthreads();
  for (int st = 128; st > 0; st >>= 1) {
    if (threadIdx.x < st) {
      sh0[threadIdx.x] += sh0[threadIdx.x + st];
      sh1[threadIdx.x] += sh1[threadIdx.x + st];
    }
    __syncthreads();
  }
  if (threadIdx.x == 0) {
    float mu  = sh0[0] / (float)TN;
    float var = sh1[0] / (float)TN - mu * mu;
    stats[c]       = mu;
    stats[128 + c] = rsqrtf(var + 1e-5f);
  }
}

// ---------------- AdaptivePointNorm apply ----------------
__global__ void __launch_bounds__(256)
ada_apply_k(const float* __restrict__ HR, const float* __restrict__ stats,
            const float* __restrict__ SA, float* __restrict__ Hf,
            __bf16* __restrict__ Hb) {
  int i = blockIdx.x * 256 + threadIdx.x;
  if (i >= TN * 128) return;
  int t = i >> 7, c = i & 127;
  float hn = (HR[i] - stats[c]) * stats[128 + c];
  float g  = SA[(size_t)t * 256 + c];
  float b  = SA[(size_t)t * 256 + 128 + c];
  float v  = g * hn + b;
  Hf[i] = v;
  Hb[i] = f2bf(v);
}

// ---------------- global max pool per batch ----------------
__global__ void __launch_bounds__(256)
segmax_k(const float* __restrict__ H, float* __restrict__ G) {
  int blk = blockIdx.x;              // 512 = 4 batches * 128 channels
  int b = blk >> 7, c = blk & 127;
  float m = -3.4e38f;
  for (int n = threadIdx.x; n < NPTS; n += 256)
    m = fmaxf(m, H[((size_t)(b * NPTS + n)) * 128 + c]);
  __shared__ float sh[256];
  sh[threadIdx.x] = m;
  __syncthreads();
  for (int st = 128; st > 0; st >>= 1) {
    if (threadIdx.x < st) sh[threadIdx.x] = fmaxf(sh[threadIdx.x], sh[threadIdx.x + st]);
    __syncthreads();
  }
  if (threadIdx.x == 0) G[blk] = sh[0];
}

// ---------------- tiny [4,128] linear + leaky(0.01) ----------------
__global__ void __launch_bounds__(256)
small_lin_k(const float* __restrict__ X, const float* __restrict__ W,
            const float* __restrict__ b, float* __restrict__ Y) {
  int tid = blockIdx.x * 256 + threadIdx.x;   // 512 total
  if (tid >= 512) return;
  int r = tid >> 7, o = tid & 127;
  float s = b[o];
  for (int c = 0; c < 128; c++) s += X[r * 128 + c] * W[o * 128 + c];
  Y[tid] = s > 0.0f ? s : 0.01f * s;
}

// ---------------- xc = [x | g[batch]] bf16 [T,256] ----------------
__global__ void __launch_bounds__(256)
pack_xc_k(const __bf16* __restrict__ Hb, const float* __restrict__ Gact,
          __bf16* __restrict__ XC) {
  int i = blockIdx.x * 256 + threadIdx.x;
  if (i >= TN * 256) return;
  int t = i >> 8, c = i & 255;
  if (c < 128) XC[i] = Hb[(size_t)t * 128 + c];
  else         XC[i] = f2bf(Gact[(t / NPTS) * 128 + (c - 128)]);
}

// ---------------- pc = 0.75 * tanh(W3 @ t2 + b3) : 64 -> 3 ----------------
__global__ void __launch_bounds__(256)
tail3_k(const __bf16* __restrict__ T2, const float* __restrict__ W,
        const float* __restrict__ b, float* __restrict__ PC) {
  int t = blockIdx.x * 256 + threadIdx.x;
  if (t >= TN) return;
  for (int d = 0; d < 3; d++) {
    float s = b[d];
    for (int c = 0; c < 64; c++)
      s += bf2f(T2[(size_t)t * 64 + c]) * W[d * 64 + c];
    PC[t * 3 + d] = 0.75f * tanhf(s);
  }
}

// ---------------- final output [B,N,131] ----------------
__global__ void __launch_bounds__(256)
final_pack_k(const float* __restrict__ H, const float* __restrict__ PC,
             float* __restrict__ out) {
  int i = blockIdx.x * 256 + threadIdx.x;
  if (i >= TN * 131) return;
  int t = i / 131, c = i - t * 131;
  out[i] = (c < 128) ? H[(size_t)t * 128 + c] : PC[t * 3 + (c - 128)];
}

// =====================================================================
extern "C" void kernel_launch(void* const* d_in, const int* in_sizes, int n_in,
                              void* d_out, int out_size, void* d_ws, size_t ws_size,
                              hipStream_t stream) {
  (void)in_sizes; (void)n_in; (void)out_size; (void)ws_size;
  const float* z    = (const float*)d_in[0];
  const float* pos  = (const float*)d_in[1];
  const float* encb = (const float*)d_in[2];
  const int*   src  = (const int*)d_in[3];      // edge_index[0]

  // param leaf indices (W then b, setup_inputs() insertion order)
  // 5,6: pe_style1  7,8: pe_style2  9..20: pe_blk0  21..32: pe_blk1
  // 33,34 gc1  35,36 gc2  37,38 t1  39,40 t2  41,42 t3
  // 43,44 gg_s1  45,46 gg_s2  47,48 gg_fe  49..60 gg_blk0

  char* ws = (char*)d_ws;
  size_t off = 0;
  auto arena = [&](size_t bytes) -> char* {
    char* p = ws + off;
    off = (off + bytes + 255) & ~(size_t)255;
    return p;
  };

  // --- converted weight arenas (bf16, K-padded) ---
  char* Wps1 = arena(128 * 160 * 2);
  char* Wps2 = arena(128 * 128 * 2);
  struct BlkW { char *h1, *f, *g1, *g2, *ada; };
  auto alloc_blk = [&]() {
    BlkW w;
    w.h1  = arena(128 * 128 * 2);
    w.f   = arena(128 * 160 * 2);
    w.g1  = arena(128 * 128 * 2);
    w.g2  = arena(128 * 128 * 2);
    w.ada = arena(256 * 128 * 2);
    return w;
  };
  BlkW Wb0 = alloc_blk(), Wb1 = alloc_blk(), Wgg = alloc_blk();
  char* Wt1  = arena(128 * 256 * 2);
  char* Wt2  = arena(64  * 128 * 2);
  char* Wgs1 = arena(128 * 160 * 2);
  char* Wgs2 = arena(128 * 128 * 2);
  char* Wfe  = arena(128 * 256 * 2);

  // --- activation buffers ---
  char* X160 = arena((size_t)TN * 160 * 2);   // bf16
  char* S1   = arena((size_t)TN * 128 * 2);   // bf16
  char* STY  = arena((size_t)TN * 128 * 2);   // bf16
  char* SA   = arena((size_t)TN * 256 * 4);   // f32
  char* Hf   = arena((size_t)TN * 128 * 4);   // f32
  char* Hb   = arena((size_t)TN * 128 * 2);   // bf16
  char* T1b  = arena((size_t)TN * 128 * 2);   // bf16
  char* T2b  = arena((size_t)TN * 64  * 2);   // bf16
  char* DLT  = arena((size_t)TN * 3   * 4);   // f32
  char* AGG  = arena((size_t)TN * 128 * 2);   // bf16
  char* G2   = arena((size_t)TN * 128 * 4);   // f32
  char* HR   = arena((size_t)TN * 128 * 4);   // f32
  char* STAT = arena(256 * 4);
  char* Gp   = arena(512 * 4);
  char* Gt   = arena(512 * 4);
  char* Ga   = arena(512 * 4);
  char* XC   = arena((size_t)TN * 256 * 2);   // bf16
  char* PC   = arena((size_t)TN * 3   * 4);   // f32

  auto cvt = [&](int widx, char* dst, int rows, int kin, int kpad) {
    int n = rows * kpad;
    cvt_w<<<(n + 255) / 256, 256, 0, stream>>>(
        (const float*)d_in[widx], (__bf16*)dst, rows, kin, kpad);
  };
  auto cvt_blk = [&](int base, BlkW& w) {
    cvt(base + 0,  w.h1,  128, 128, 128);
    cvt_wf<<<(128 * 160 + 255) / 256, 256, 0, stream>>>(
        (const float*)d_in[base + 4], (__bf16*)w.f);
    cvt(base + 6,  w.g1,  128, 128, 128);
    cvt(base + 8,  w.g2,  128, 128, 128);
    cvt(base + 10, w.ada, 256, 128, 128);
  };
  cvt(5, Wps1, 128, 131, 160);
  cvt(7, Wps2, 128, 128, 128);
  cvt_blk(9,  Wb0);
  cvt_blk(21, Wb1);
  cvt(37, Wt1, 128, 256, 256);
  cvt(39, Wt2, 64, 128, 128);
  cvt(43, Wgs1, 128, 131, 160);
  cvt(45, Wgs2, 128, 128, 128);
  cvt(47, Wfe, 128, 256, 256);
  cvt_blk(49, Wgg);

  auto gemm = [&](const char* X, int ldx, const char* W, int bidx,
                  char* Yf, char* Yb, int Nout, int act) {
    int tiles = (TN / 16) * (Nout / 64);
    gemm_wmma<<<tiles / 8, 256, 0, stream>>>(
        (const __bf16*)X, ldx, (const __bf16*)W,
        bidx >= 0 ? (const float*)d_in[bidx] : nullptr,
        (float*)Yf, (__bf16*)Yb, Nout, act);
  };

  auto run_block = [&](int base, BlkW& w, const float* posp) {
    // delta = tanh(h2(relu(h1(h))))
    gemm(Hb, 128, w.h1, base + 1, nullptr, T1b, 128, 1);
    delta_k<<<TN / 256, 256, 0, stream>>>(
        (const __bf16*)T1b, (const float*)d_in[base + 2],
        (const float*)d_in[base + 3], (float*)DLT);
    // style affine params
    gemm(STY, 128, w.ada, base + 11, SA, nullptr, 256, 0);
    // edge messages + segment-sum
    edge_gemm<<<TN / 16, 256, 0, stream>>>(
        (const __bf16*)Hb, posp, (const float*)DLT, src,
        (const __bf16*)w.f, (const float*)d_in[base + 5], (__bf16*)AGG);
    // g2(relu(g1(agg)))
    gemm(AGG, 128, w.g1, base + 7, nullptr, T1b, 128, 1);
    gemm(T1b, 128, w.g2, base + 9, G2, nullptr, 128, 0);
    // residual + leaky(0.2), instance-norm stats, AdaPN apply
    residual_k<<<TN * 128 / 256, 256, 0, stream>>>(
        (const float*)Hf, (const float*)G2, (float*)HR);
    colstats_k<<<128, 256, 0, stream>>>((const float*)HR, (float*)STAT);
    ada_apply_k<<<TN * 128 / 256, 256, 0, stream>>>(
        (const float*)HR, (const float*)STAT, (const float*)SA,
        (float*)Hf, (__bf16*)Hb);
  };

  // ---------- CloudGenerator ----------
  pack160<<<TN * 160 / 256, 256, 0, stream>>>(z, pos, (__bf16*)X160);
  gemm(X160, 160, Wps1, 6, nullptr, S1, 128, 2);
  gemm(S1,   128, Wps2, 8, nullptr, STY, 128, 2);
  rff_k<<<TN * 64 / 256, 256, 0, stream>>>(pos, encb, (float*)Hf, (__bf16*)Hb);
  run_block(9,  Wb0, pos);
  run_block(21, Wb1, pos);
  segmax_k<<<512, 256, 0, stream>>>((const float*)Hf, (float*)Gp);
  small_lin_k<<<2, 256, 0, stream>>>((const float*)Gp,
      (const float*)d_in[33], (const float*)d_in[34], (float*)Gt);
  small_lin_k<<<2, 256, 0, stream>>>((const float*)Gt,
      (const float*)d_in[35], (const float*)d_in[36], (float*)Ga);
  pack_xc_k<<<TN * 256 / 256, 256, 0, stream>>>(
      (const __bf16*)Hb, (const float*)Ga, (__bf16*)XC);
  gemm(XC,  256, Wt1, 38, nullptr, T1b, 128, 2);
  gemm(T1b, 128, Wt2, 40, nullptr, T2b, 64, 2);
  tail3_k<<<TN / 256, 256, 0, stream>>>(
      (const __bf16*)T2b, (const float*)d_in[41], (const float*)d_in[42],
      (float*)PC);

  // ---------- GaussiansGenerator ----------
  pack160<<<TN * 160 / 256, 256, 0, stream>>>(z, (const float*)PC, (__bf16*)X160);
  gemm(X160, 160, Wgs1, 44, nullptr, S1, 128, 2);
  gemm(S1,   128, Wgs2, 46, nullptr, STY, 128, 2);
  gemm(XC,   256, Wfe,  48, Hf, Hb, 128, 2);
  run_block(49, Wgg, (const float*)PC);

  final_pack_k<<<TN * 131 / 256, 256, 0, stream>>>(
      (const float*)Hf, (const float*)PC, (float*)d_out);
}
